// Hedgehog_17076789969437
// MI455X (gfx1250) — compile-verified, matrix operand order checked by reading, not run on hardware
//
#include <hip/hip_runtime.h>
#include <hip/hip_bf16.h>
#include <math.h>

// ---------------------------------------------------------------------------
// Hedgehog causal linear attention, chunked scan, CDNA5 WMMA (bf16 -> f32)
// with async global->LDS staging (double-buffered software pipeline).
//   b=1, h=16, l=1024, d=64, f=64, D=2f=128, dv=64 (padded to 80 w/ ones col)
// One block per head (16 blocks, 256 threads = 8 wave32).
// ---------------------------------------------------------------------------

typedef __attribute__((ext_vector_type(16))) __bf16 v16bf;
typedef __attribute__((ext_vector_type(8)))  float  v8f;
typedef __attribute__((ext_vector_type(4)))  int    v4i;

#define NH     16
#define SEQ    1024
#define DK     64      // model dim per head (q/k/v inner)
#define FD     64      // feature dim per softmax branch
#define DM     128     // 2*FD  (concat +softmax / -softmax)
#define DVP    80      // dv(64) + ones col(1), padded to 5 tiles of 16
#define CHUNK  64
#define NCHUNK (SEQ / CHUNK)
#define EPSF   1e-12f
#define NTHR   256

// Async global->LDS staging (CDNA5 GLOBAL_LOAD_ASYNC_TO_LDS, ASYNCcnt).
// Builtin signature (from hipcc diagnostics): (int4 AS1*, int4 AS3*, imm, imm)
#if defined(__AMDGCN__) && __has_builtin(__builtin_amdgcn_global_load_async_to_lds_b128) && __has_builtin(__builtin_amdgcn_s_wait_asynccnt)
#define USE_ASYNC 1
typedef __attribute__((address_space(1))) v4i gv4i;   // global-memory int4
typedef __attribute__((address_space(3))) v4i lv4i;   // LDS int4
#else
#define USE_ASYNC 0
#endif

// LDS carve (bytes)
#define OFF_WQT   0                                  // [FD][DK] bf16   8KB
#define OFF_WKT   (OFF_WQT + FD*DK*2)                // [FD][DK] bf16   8KB
#define OFF_X     (OFF_WKT + FD*DK*2)                // [CHUNK][DK] bf16 8KB
#define OFF_QF    (OFF_X   + CHUNK*DK*2)             // [CHUNK][DM] bf16 16KB
#define OFF_KF    (OFF_QF  + CHUNK*DM*2)             // [CHUNK][DM] bf16 16KB
#define OFF_KFT   (OFF_KF  + CHUNK*DM*2)             // [DM][CHUNK] bf16 16KB
#define OFF_AM    (OFF_KFT + DM*CHUNK*2)             // [CHUNK][CHUNK] bf16 8KB
#define OFF_VT    (OFF_AM  + CHUNK*CHUNK*2)          // [DVP][CHUNK] bf16 10KB
#define OFF_SB    (OFF_VT  + DVP*CHUNK*2)            // [DVP][DM] bf16  20KB
#define OFF_H     (OFF_SB  + DVP*DM*2)               // [CHUNK][FD] f32 16KB
#define OFF_Z     (OFF_H   + CHUNK*FD*4)             // [CHUNK] f32
#define OFF_QS    (OFF_Z   + CHUNK*4)                // [CHUNK][DK] f32 16KB staging
#define OFF_KS    (OFF_QS  + CHUNK*DK*4)             // [CHUNK][DK] f32 16KB staging
#define OFF_VS    (OFF_KS  + CHUNK*DK*4)             // [CHUNK][DK] f32 16KB staging
#define SMEM_BYTES (OFF_VS + CHUNK*DK*4)             // ~175KB (< 320KB WGP LDS)

// Load a 16x32 bf16 WMMA fragment from an LDS matrix.
// A-role: buffer is row-major [M][K]; B-role: buffer is the TRANSPOSED [N][K]
// storage. Uses contiguous k order, same for A and B (lane<16 -> K 0..15,
// lane>=16 -> K 16..31; VGPR j = K 2j,2j+1), each lane reads 16 contiguous
// bf16 -> two ds_load_b128 per fragment.
__device__ __forceinline__ v16bf ldfrag(const __bf16* base, int rowbase, int ld, int ks) {
  const int lane = threadIdx.x & 31;
  const __bf16* p = base + (rowbase + (lane & 15)) * ld + ks + ((lane >> 4) << 4);
  v16bf r;
#pragma unroll
  for (int j = 0; j < 16; ++j) r[j] = p[j];
  return r;
}

__device__ __forceinline__ v8f wmma_bf(v16bf a, v16bf b, v8f c) {
  return __builtin_amdgcn_wmma_f32_16x16x32_bf16(
      /*neg_a=*/false, a, /*neg_b=*/false, b,
      /*c_mod=*/(short)0, c, /*reuse_a=*/false, /*reuse_b=*/false);
}

// Stage one chunk (q,k,v; CHUNK*DK fp32 each) from global into LDS staging.
// Async path: per-lane 16B GLOBAL_LOAD_ASYNC_TO_LDS_B128, 4 block-wide issues
// per buffer. Fallback: synchronous float4 copy (same data flow).
__device__ __forceinline__ void stage_chunk(const float* qg, const float* kg,
                                            const float* vg, float* qs,
                                            float* ks, float* vs) {
  const int tid = threadIdx.x;
#if USE_ASYNC
#pragma unroll
  for (int it = 0; it < (CHUNK * DK / 4) / NTHR; ++it) {
    int idx = it * NTHR + tid;
    __builtin_amdgcn_global_load_async_to_lds_b128(
        (gv4i*)(qg + idx * 4), (lv4i*)(qs + idx * 4), 0, 0);
    __builtin_amdgcn_global_load_async_to_lds_b128(
        (gv4i*)(kg + idx * 4), (lv4i*)(ks + idx * 4), 0, 0);
    __builtin_amdgcn_global_load_async_to_lds_b128(
        (gv4i*)(vg + idx * 4), (lv4i*)(vs + idx * 4), 0, 0);
  }
#else
  const float4* qg4 = (const float4*)qg;
  const float4* kg4 = (const float4*)kg;
  const float4* vg4 = (const float4*)vg;
  float4* qs4 = (float4*)qs;
  float4* ks4 = (float4*)ks;
  float4* vs4 = (float4*)vs;
#pragma unroll
  for (int it = 0; it < (CHUNK * DK / 4) / NTHR; ++it) {
    int idx = it * NTHR + tid;
    qs4[idx] = qg4[idx];
    ks4[idx] = kg4[idx];
    vs4[idx] = vg4[idx];
  }
#endif
}

__device__ __forceinline__ void wait_staging() {
#if USE_ASYNC
  __builtin_amdgcn_s_wait_asynccnt(0);
#endif
}

// Parallel row softmax: 4 lanes per row (16 elems each), wave32 shfl_xor
// reductions. Writes clamp(concat(softmax(h), softmax(-h)), eps)*sc to
// rowout [row][DM] and optionally to the transposed colT [DM][row].
__device__ __forceinline__ void softmax_rows(const float* sH, __bf16* rowout,
                                             __bf16* colT, float sc) {
  const int tid  = threadIdx.x;
  const int row  = tid >> 2;
  const int part = tid & 3;
  const float* h = sH + row * FD + part * 16;
  float mx = -1e30f, mn = 1e30f;
#pragma unroll
  for (int j = 0; j < 16; ++j) { float x = h[j]; mx = fmaxf(mx, x); mn = fminf(mn, x); }
  mx = fmaxf(mx, __shfl_xor(mx, 1)); mx = fmaxf(mx, __shfl_xor(mx, 2));
  mn = fminf(mn, __shfl_xor(mn, 1)); mn = fminf(mn, __shfl_xor(mn, 2));
  float s1 = 0.f, s2 = 0.f;
#pragma unroll
  for (int j = 0; j < 16; ++j) { s1 += __expf(h[j] - mx); s2 += __expf(mn - h[j]); }
  s1 += __shfl_xor(s1, 1); s1 += __shfl_xor(s1, 2);
  s2 += __shfl_xor(s2, 1); s2 += __shfl_xor(s2, 2);
  const float r1 = 1.f / s1, r2 = 1.f / s2;
  __bf16* ro = rowout + row * DM + part * 16;
#pragma unroll
  for (int j = 0; j < 16; ++j) {
    int col = part * 16 + j;
    float a = fmaxf(__expf(h[j] - mx) * r1, EPSF) * sc;
    float b = fmaxf(__expf(mn - h[j]) * r2, EPSF) * sc;
    ro[j]      = (__bf16)a;
    ro[FD + j] = (__bf16)b;
    if (colT) {
      colT[col * CHUNK + row]        = (__bf16)a;
      colT[(FD + col) * CHUNK + row] = (__bf16)b;
    }
  }
}

__global__ void hedgehog_chunked_wmma(const float* __restrict__ q,
                                      const float* __restrict__ k,
                                      const float* __restrict__ v,
                                      const float* __restrict__ Wq,
                                      const float* __restrict__ Wk,
                                      const int* __restrict__ use_scale,
                                      const int* __restrict__ use_norm,
                                      float* __restrict__ out) {
  extern __shared__ char smem[];
  __bf16* sWqT = (__bf16*)(smem + OFF_WQT);
  __bf16* sWkT = (__bf16*)(smem + OFF_WKT);
  __bf16* sX   = (__bf16*)(smem + OFF_X);
  __bf16* sQF  = (__bf16*)(smem + OFF_QF);
  __bf16* sKF  = (__bf16*)(smem + OFF_KF);
  __bf16* sKFT = (__bf16*)(smem + OFF_KFT);
  __bf16* sAM  = (__bf16*)(smem + OFF_AM);
  __bf16* sVT  = (__bf16*)(smem + OFF_VT);
  __bf16* sSb  = (__bf16*)(smem + OFF_SB);
  float*  sH   = (float*)(smem + OFF_H);
  float*  sZ   = (float*)(smem + OFF_Z);
  float*  sQs  = (float*)(smem + OFF_QS);
  float*  sKs  = (float*)(smem + OFF_KS);
  float*  sVs  = (float*)(smem + OFF_VS);

  const int head = blockIdx.x;
  const int tid  = threadIdx.x;
  const int wave = tid >> 5;           // 0..7
  const int lane = tid & 31;
  const int l15  = lane & 15;
  const int lh8  = (lane >> 4) << 3;   // C/D layout: row = 16*mt + v + lh8

  const bool do_norm = use_norm[0] != 0;
  const float qscale = (use_scale[0] != 0) ? rsqrtf((float)DM) : 1.0f;

  const float* qh = q + (size_t)head * SEQ * DK;
  const float* kh = k + (size_t)head * SEQ * DK;
  const float* vh = v + (size_t)head * SEQ * DK;

  // --- load Wq^T, Wk^T (global [h][d][f] -> LDS [f][d]) ---
  for (int i = tid; i < FD * DK; i += NTHR) {
    int f = i / DK, d = i % DK;
    sWqT[f * DK + d] = (__bf16)Wq[((size_t)head * DK + d) * FD + f];
    sWkT[f * DK + d] = (__bf16)Wk[((size_t)head * DK + d) * FD + f];
  }

  // Kick off staging for chunk 0.
  stage_chunk(qh, kh, vh, sQs, sKs, sVs);

  // Persistent state S^T [DVP][DM] in registers: wave w owns D-strip
  // n-tile = w (cols w*16..w*16+15), 5 m-tiles (rows 0..79). 40 VGPRs/thread.
  v8f S[5];
#pragma unroll
  for (int i = 0; i < 5; ++i) S[i] = (v8f){0.f,0.f,0.f,0.f,0.f,0.f,0.f,0.f};

  for (int c = 0; c < NCHUNK; ++c) {
    const int r0 = c * CHUNK;

    // (1) stage S_prev as bf16 into LDS (B-role transposed storage [n][D])
#pragma unroll
    for (int i = 0; i < 5; ++i)
#pragma unroll
      for (int vv = 0; vv < 8; ++vv)
        sSb[(i * 16 + vv + lh8) * DM + wave * 16 + l15] = (__bf16)S[i][vv];

    // staging for this chunk complete + make S_prev / staging visible
    wait_staging();
    __syncthreads();

    // (2) convert q staging -> bf16 row-major; build v^T with ones column
    for (int i = tid; i < CHUNK * DK; i += NTHR)
      sX[i] = (__bf16)sQs[i];
    for (int i = tid; i < DVP * CHUNK; i += NTHR) {
      int n = i / CHUNK, s = i % CHUNK;
      float val = (n < DK) ? sVs[s * DK + n] : (n == DK ? 1.0f : 0.0f);
      sVT[i] = (__bf16)val;
    }
    __syncthreads();

    // (3) hq = q_chunk @ Wq  (4x4 tiles of 16x16, K=64)
    for (int t = wave; t < 16; t += 8) {
      int mt = t >> 2, nt = t & 3;
      v8f hc = (v8f){0.f,0.f,0.f,0.f,0.f,0.f,0.f,0.f};
      hc = wmma_bf(ldfrag(sX, mt * 16, DK, 0),  ldfrag(sWqT, nt * 16, DK, 0),  hc);
      hc = wmma_bf(ldfrag(sX, mt * 16, DK, 32), ldfrag(sWqT, nt * 16, DK, 32), hc);
#pragma unroll
      for (int vv = 0; vv < 8; ++vv)
        sH[(mt * 16 + vv + lh8) * FD + nt * 16 + l15] = hc[vv];
    }
    __syncthreads();

    // (4) qf = clamp(concat(softmax(h), softmax(-h)), eps) * D^-0.5
    softmax_rows(sH, sQF, nullptr, qscale);
    __syncthreads();

    // (5) same for k: convert staging, hk = k_chunk @ Wk, softmax -> kf, kf^T
    for (int i = tid; i < CHUNK * DK; i += NTHR)
      sX[i] = (__bf16)sKs[i];
    __syncthreads();

    // All staging consumed -> prefetch next chunk asynchronously; the copy
    // overlaps the WMMA-heavy phase below (steps 5b-8).
    if (c + 1 < NCHUNK) {
      const int rn = (c + 1) * CHUNK;
      stage_chunk(qh + (size_t)rn * DK, kh + (size_t)rn * DK,
                  vh + (size_t)rn * DK, sQs, sKs, sVs);
    }

    for (int t = wave; t < 16; t += 8) {
      int mt = t >> 2, nt = t & 3;
      v8f hc = (v8f){0.f,0.f,0.f,0.f,0.f,0.f,0.f,0.f};
      hc = wmma_bf(ldfrag(sX, mt * 16, DK, 0),  ldfrag(sWkT, nt * 16, DK, 0),  hc);
      hc = wmma_bf(ldfrag(sX, mt * 16, DK, 32), ldfrag(sWkT, nt * 16, DK, 32), hc);
#pragma unroll
      for (int vv = 0; vv < 8; ++vv)
        sH[(mt * 16 + vv + lh8) * FD + nt * 16 + l15] = hc[vv];
    }
    __syncthreads();
    softmax_rows(sH, sKF, sKFT, 1.0f);
    __syncthreads();

    // (6) A = causal_mask(qf @ kf^T), K = DM = 128 (4 WMMAs/tile), store bf16
    for (int t = wave; t < 16; t += 8) {
      int mt = t >> 2, nt = t & 3;
      v8f acc = (v8f){0.f,0.f,0.f,0.f,0.f,0.f,0.f,0.f};
#pragma unroll
      for (int ks = 0; ks < DM; ks += 32)
        acc = wmma_bf(ldfrag(sQF, mt * 16, DM, ks), ldfrag(sKF, nt * 16, DM, ks), acc);
#pragma unroll
      for (int vv = 0; vv < 8; ++vv) {
        int rr = mt * 16 + vv + lh8, ss = nt * 16 + l15;
        sAM[rr * CHUNK + ss] = (ss <= rr) ? (__bf16)acc[vv] : (__bf16)0.0f;
      }
    }
    __syncthreads();

    // (7a) normalizer tiles: output cols 64..79 (col 64 = z before eps).
    //      o' = qf @ S_prev^T-strip + A_mask @ v'^T-strip
    if (wave < 4) {
      int mt = wave;
      v8f acc = (v8f){0.f,0.f,0.f,0.f,0.f,0.f,0.f,0.f};
#pragma unroll
      for (int ks = 0; ks < DM; ks += 32)
        acc = wmma_bf(ldfrag(sQF, mt * 16, DM, ks), ldfrag(sSb, 64, DM, ks), acc);
#pragma unroll
      for (int ks = 0; ks < CHUNK; ks += 32)
        acc = wmma_bf(ldfrag(sAM, mt * 16, CHUNK, ks), ldfrag(sVT, 64, CHUNK, ks), acc);
      if (l15 == 0) {
#pragma unroll
        for (int vv = 0; vv < 8; ++vv)
          sZ[mt * 16 + vv + lh8] = acc[vv] + EPSF;
      }
    }
    __syncthreads();

    // (7b) output tiles (cols 0..63): o = (qf@S_prev + A_mask@v) / z
    {
      float* og = out + ((size_t)head * SEQ + r0) * DK;
      for (int t = wave; t < 16; t += 8) {
        int mt = t >> 2, nt = t & 3;
        v8f acc = (v8f){0.f,0.f,0.f,0.f,0.f,0.f,0.f,0.f};
#pragma unroll
        for (int ks = 0; ks < DM; ks += 32)
          acc = wmma_bf(ldfrag(sQF, mt * 16, DM, ks), ldfrag(sSb, nt * 16, DM, ks), acc);
#pragma unroll
        for (int ks = 0; ks < CHUNK; ks += 32)
          acc = wmma_bf(ldfrag(sAM, mt * 16, CHUNK, ks), ldfrag(sVT, nt * 16, CHUNK, ks), acc);
#pragma unroll
        for (int vv = 0; vv < 8; ++vv) {
          int rr = mt * 16 + vv + lh8;
          float val = acc[vv];
          if (do_norm) val /= sZ[rr];
          og[rr * DK + nt * 16 + l15] = val;
        }
      }
    }

    // (8) state update: S^T += v'^T @ kf  (A = v'^T [80x64], B^T-storage = kf^T)
#pragma unroll
    for (int i = 0; i < 5; ++i) {
      S[i] = wmma_bf(ldfrag(sVT, i * 16, CHUNK, 0),  ldfrag(sKFT, wave * 16, CHUNK, 0),  S[i]);
      S[i] = wmma_bf(ldfrag(sVT, i * 16, CHUNK, 32), ldfrag(sKFT, wave * 16, CHUNK, 32), S[i]);
    }
    __syncthreads();   // protect sSb/sVT/sKF/sX against next-chunk overwrite
  }
}

extern "C" void kernel_launch(void* const* d_in, const int* in_sizes, int n_in,
                              void* d_out, int out_size, void* d_ws, size_t ws_size,
                              hipStream_t stream) {
  const float* q  = (const float*)d_in[0];
  const float* k  = (const float*)d_in[1];
  const float* v  = (const float*)d_in[2];
  const float* Wq = (const float*)d_in[3];
  const float* Wk = (const float*)d_in[4];
  const int* use_scale = (const int*)d_in[5];
  const int* use_norm  = (const int*)d_in[6];
  float* out = (float*)d_out;
  (void)in_sizes; (void)n_in; (void)out_size; (void)d_ws; (void)ws_size;

  hedgehog_chunked_wmma<<<dim3(NH), dim3(NTHR), SMEM_BYTES, stream>>>(
      q, k, v, Wq, Wk, use_scale, use_norm, out);
}
